// RNN_30580167147721
// MI455X (gfx1250) — matrix operand order repaired and checked by reading, not compiled, not run on hardware
//
#include <hip/hip_runtime.h>
#include <math.h>

typedef __attribute__((ext_vector_type(16))) __bf16 v16bf;
typedef __attribute__((ext_vector_type(8)))  __bf16 v8bf;
typedef __attribute__((ext_vector_type(8)))  float  v8f;
typedef __attribute__((ext_vector_type(4)))  float  f32x4;

namespace {
constexpr int kT = 128, kB = 64, kE = 1024, kH = 1024, kV = 10000, kL = 2;
}

// ---------------------------------------------------------------------------
// Elementwise f32 -> bf16 weight preconversion (n must be a multiple of 8).
// ---------------------------------------------------------------------------
__global__ __launch_bounds__(256)
void cvt_f32_bf16(const float* __restrict__ src, __bf16* __restrict__ dst, int n)
{
  int i = (blockIdx.x * blockDim.x + threadIdx.x) * 8;
  if (i >= n) return;
  f32x4 a = *(const f32x4*)(src + i);
  f32x4 b = *(const f32x4*)(src + i + 4);
  v8bf o;
  #pragma unroll
  for (int k = 0; k < 4; ++k) {
    o[k]     = (__bf16)a[k];
    o[4 + k] = (__bf16)b[k];
  }
  *(v8bf*)(dst + i) = o;
}

// ---------------------------------------------------------------------------
// Wave-level GEMM: Y = act(X @ W.T + bias + addRow[row%B])
//   X: (M,K) f32 (row source per xMode), W: (N,K) bf16 row-major (precvt).
// Each wave computes 32 rows x 64 cols = 2 A-frags x 4 B-tiles = 8 WMMA/K-step.
// Block = 8 waves as 4(M) x 2(N): 128 rows x 128 cols per block.
// Inner K-loop is branch-free; N-edge handled by clamped W rows + store guard.
// ---------------------------------------------------------------------------
__global__ __launch_bounds__(256)
void wgemm_bf16(const float* __restrict__ X, const int* __restrict__ rowIdx,
                int xMode, int lIn, int tOffIn,
                const __bf16* __restrict__ W, const float* __restrict__ bias,
                const float* __restrict__ addRow,
                float* __restrict__ Y, int outMode, int lOut, int tOffOut,
                int M, int N, int K, int doTanh)
{
  const int lane   = threadIdx.x & 31;
  const int waveId = threadIdx.x >> 5;
  const int waveM  = waveId >> 1;              // 0..3
  const int waveN  = waveId & 1;               // 0..1
  const int row0   = blockIdx.y * 128 + waveM * 32;
  const int nBase  = blockIdx.x * 128 + waveN * 64;
  if (row0 >= M || nBase >= N) return;         // wave-uniform
  const int cIn   = lane & 15;
  const int half  = lane >> 4;                 // 0 or 1

  // ---- A row pointers for this lane (2 row-fragments: row0+cIn, row0+16+cIn)
  const float* Xrow[2];
  #pragma unroll
  for (int m = 0; m < 2; ++m) {
    const int r = row0 + 16 * m + cIn;
    if (xMode == 0) {
      Xrow[m] = X + (size_t)r * (size_t)K;                  // linear
    } else if (xMode == 1) {
      Xrow[m] = X + (size_t)rowIdx[r] * (size_t)K;          // embedding gather
    } else {
      const int t = r / kB + tOffIn;                        // h_all strided
      const int b = r & (kB - 1);
      Xrow[m] = X + ((size_t)(t * kL + lIn) * kB + b) * (size_t)K;
    }
  }

  // ---- B row pointers per tile (clamped in-bounds for N edge) ----
  const __bf16* Wrow[4];
  #pragma unroll
  for (int j = 0; j < 4; ++j) {
    const int c = nBase + 16 * j + cIn;
    Wrow[j] = W + (size_t)(c < N ? c : 0) * (size_t)K;
  }

  v8f acc[2][4] = {};

  const int kA  = half * 16;   // A lane K-base offset within a 32-chunk
  const int kBo = half * 16;   // B lane K-base offset within a 32-chunk

  for (int kb = 0; kb < K; kb += 32) {
    // B fragments: 16 contiguous bf16 per lane per tile (32B direct load)
    v16bf bfr[4];
    #pragma unroll
    for (int j = 0; j < 4; ++j)
      bfr[j] = *(const v16bf*)(Wrow[j] + kb + kBo);

    // A fragments: elems 0..15 = K kb+kA+0..15
    v16bf a[2];
    #pragma unroll
    for (int m = 0; m < 2; ++m) {
      const float* pA = Xrow[m] + kb + kA;
      f32x4 u0 = *(const f32x4*)(pA);
      f32x4 u1 = *(const f32x4*)(pA + 4);
      f32x4 u2 = *(const f32x4*)(pA + 8);
      f32x4 u3 = *(const f32x4*)(pA + 12);
      #pragma unroll
      for (int i = 0; i < 4; ++i) {
        a[m][i]      = (__bf16)u0[i];
        a[m][4 + i]  = (__bf16)u1[i];
        a[m][8 + i]  = (__bf16)u2[i];
        a[m][12 + i] = (__bf16)u3[i];
      }
    }

    #pragma unroll
    for (int m = 0; m < 2; ++m)
      #pragma unroll
      for (int j = 0; j < 4; ++j)
        acc[m][j] = __builtin_amdgcn_wmma_f32_16x16x32_bf16(
            false, a[m], false, bfr[j], (short)0, acc[m][j], false, false);
  }

  // ---- epilogue: bias + addRow + tanh + store (N-edge guarded here) ----
  int numJ = (N - nBase) >> 4; if (numJ > 4) numJ = 4;      // uniform
  #pragma unroll
  for (int m = 0; m < 2; ++m) {
    #pragma unroll
    for (int j = 0; j < 4; ++j) {
      if (j >= numJ) continue;
      const int   c  = nBase + 16 * j + cIn;
      const float bv = bias ? bias[c] : 0.0f;
      #pragma unroll
      for (int i = 0; i < 8; ++i) {
        const int row = row0 + 16 * m + half * 8 + i;
        float v = acc[m][j][i] + bv;
        if (addRow) v += addRow[(size_t)(row & (kB - 1)) * (size_t)N + c];
        if (doTanh) v = tanhf(v);
        size_t off;
        if (outMode == 0) {
          off = (size_t)row * (size_t)N + c;
        } else {
          const int t = row / kB + tOffOut;
          const int b = row & (kB - 1);
          off = ((size_t)(t * kL + lOut) * kB + b) * (size_t)N + c;
        }
        Y[off] = v;
      }
    }
  }
}

static void G(hipStream_t s,
              const float* X, const int* rowIdx, int xMode, int lIn, int tOffIn,
              const __bf16* W, const float* bias, const float* addRow,
              float* Y, int outMode, int lOut, int tOffOut,
              int M, int N, int K, int doTanh)
{
  dim3 grid((N + 127) / 128, (M + 127) / 128);
  wgemm_bf16<<<grid, dim3(256), 0, s>>>(X, rowIdx, xMode, lIn, tOffIn,
                                        W, bias, addRow, Y, outMode, lOut, tOffOut,
                                        M, N, K, doTanh);
}

static void CVT(hipStream_t s, const float* src, __bf16* dst, int n)
{
  cvt_f32_bf16<<<dim3((n / 8 + 255) / 256), dim3(256), 0, s>>>(src, dst, n);
}

extern "C" void kernel_launch(void* const* d_in, const int* in_sizes, int n_in,
                              void* d_out, int out_size, void* d_ws, size_t ws_size,
                              hipStream_t stream) {
  (void)in_sizes; (void)n_in; (void)out_size; (void)ws_size;
  const int*   inputs = (const int*)  d_in[0];   // (T,B) int32
  const float* hidden = (const float*)d_in[1];   // (L,B,H)
  const float* emb    = (const float*)d_in[2];   // (V,E)
  const float* Ww     = (const float*)d_in[3];   // (L,H,E)
  const float* Wb     = (const float*)d_in[4];   // (L,H)
  const float* Uw     = (const float*)d_in[5];   // (L,H,H)
  const float* Ub     = (const float*)d_in[6];   // (L,H)
  const float* Fw     = (const float*)d_in[7];   // (L,E,H)
  const float* Fb     = (const float*)d_in[8];   // (L,E)
  const float* Dw     = (const float*)d_in[9];   // (V,H)
  const float* Db     = (const float*)d_in[10];  // (V)
  float* out  = (float*)d_out;
  float* hOut = out + (size_t)kT * kB * kV;      // h_all region: (T*L, B, H)

  // ---- workspace layout ----
  float*  aU   = (float*)d_ws;                             // 2*B*H f32
  float*  xbuf = aU + 2 * kB * kH;                         // (T-1)*B*E f32
  __bf16* WwB  = (__bf16*)(xbuf + (size_t)(kT - 1) * kB * kE);
  __bf16* UwB  = WwB + (size_t)kL * kH * kE;
  __bf16* FwB  = UwB + (size_t)kL * kH * kH;
  __bf16* DwB  = FwB + (size_t)kL * kE * kH;

  // ---- preconvert weights to bf16 (once per launch; L2-resident after) ----
  CVT(stream, Ww, WwB, kL * kH * kE);
  CVT(stream, Uw, UwB, kL * kH * kH);
  CVT(stream, Fw, FwB, kL * kE * kH);
  CVT(stream, Dw, DwB, kV * kH);

  const int MR = (kT - 1) * kB;                  // 8128 parallel rows

  // ---- timestep 0 chain (M = 64) ----
  G(stream, hidden,           nullptr, 0, 0, 0, UwB,          Ub,      nullptr,    aU,           0, 0, 0, kB, kH, kH, 0); // aU_init l0
  G(stream, hidden + kB * kH, nullptr, 0, 0, 0, UwB + kH*kH,  Ub + kH, nullptr,    aU + kB * kH, 0, 0, 0, kB, kH, kH, 0); // aU_init l1
  G(stream, emb,  inputs,  1, 0, 0,             WwB,          Wb,      aU,         hOut, 1, 0, 0, kB, kH, kE, 1);         // h0_0 -> h_all[0,0]
  G(stream, hOut, nullptr, 2, 0, 0,             FwB,          Fb,      nullptr,    xbuf, 0, 0, 0, kB, kE, kH, 1);         // x after l0
  G(stream, xbuf, nullptr, 0, 0, 0,             WwB + kH*kE,  Wb + kH, aU + kB*kH, hOut, 1, 1, 0, kB, kH, kE, 1);         // h0_1 -> h_all[0,1]
  G(stream, hOut, nullptr, 2, 1, 0,             FwB + kE*kH,  Fb + kE, nullptr,    xbuf, 0, 0, 0, kB, kE, kH, 1);         // x after l1
  G(stream, xbuf, nullptr, 0, 0, 0,             DwB,          Db,      nullptr,    out,  0, 0, 0, kB, kV, kH, 0);         // logits[0]

  // ---- aU[l] = h0[l] @ Uw[l].T + Ub[l] (used by all t >= 1) ----
  G(stream, hOut, nullptr, 2, 0, 0, UwB,         Ub,      nullptr, aU,           0, 0, 0, kB, kH, kH, 0);
  G(stream, hOut, nullptr, 2, 1, 0, UwB + kH*kH, Ub + kH, nullptr, aU + kB * kH, 0, 0, 0, kB, kH, kH, 0);

  // ---- timesteps 1..T-1 (fully parallel, M = 8128) ----
  G(stream, emb,  inputs + kB, 1, 0, 0, WwB,         Wb,      aU,         hOut, 1, 0, 1, MR, kH, kE, 1);                  // h[t,0]
  G(stream, hOut, nullptr,     2, 0, 1, FwB,         Fb,      nullptr,    xbuf, 0, 0, 0, MR, kE, kH, 1);                  // x l0
  G(stream, xbuf, nullptr,     0, 0, 0, WwB + kH*kE, Wb + kH, aU + kB*kH, hOut, 1, 1, 1, MR, kH, kE, 1);                  // h[t,1]
  G(stream, hOut, nullptr,     2, 1, 1, FwB + kE*kH, Fb + kE, nullptr,    xbuf, 0, 0, 0, MR, kE, kH, 1);                  // x l1
  G(stream, xbuf, nullptr,     0, 0, 0, DwB,         Db,      nullptr,    out + (size_t)kB * kV, 0, 0, 0, MR, kV, kH, 0); // logits[1:]
}
